// GAT_23038204575792
// MI455X (gfx1250) — matrix operand order repaired and checked by reading, not compiled, not run on hardware
//
#include <hip/hip_runtime.h>
#include <hip/hip_bf16.h>

#define NHEADS 4
#define HIDF 32
#define DMODEL 128
#define NEG_SLOPE 0.2f

typedef __attribute__((ext_vector_type(16))) __bf16 v16bf;
typedef __attribute__((ext_vector_type(8)))  float  v8f;

// Monotone uint encoding of float so unsigned atomicMax == float max.
__device__ __forceinline__ unsigned fenc(float f) {
    unsigned u = __float_as_uint(f);
    return (u & 0x80000000u) ? ~u : (u | 0x80000000u);
}
__device__ __forceinline__ float fdec(unsigned e) {
    return (e & 0x80000000u) ? __uint_as_float(e ^ 0x80000000u)
                             : __uint_as_float(~e);
}

// ---------------------------------------------------------------------------
// Pack B (128 x NCOLS, f32 row-major) into bf16 WMMA-fragment order, zero-
// padded to NTC*16 columns.  Element order: frag = (kk32*NTC + t), then
// lane (0..31), then e (0..15), so a lane's 16 bf16 fragment values are a
// contiguous 32-byte run.  B-fragment layout per CDNA5 ISA 7.12.2:
//   lane L holds col t*16 + L%16;  K = kk32*32 + (L>=16?8:0) + (e&7) + (e>>3)*16
// ---------------------------------------------------------------------------
template <int NCOLS>
__global__ void pack_B(const float* __restrict__ B, __bf16* __restrict__ P) {
    constexpr int NTC = (NCOLS + 15) / 16;
    constexpr int TOT = 4 * NTC * 32 * 16;
    int i = blockIdx.x * blockDim.x + threadIdx.x;
    if (i >= TOT) return;
    int e    = i & 15;
    int lane = (i >> 4) & 31;
    int rest = i >> 9;
    int t    = rest % NTC;
    int kk32 = rest / NTC;
    int half = lane >> 4, lr = lane & 15;
    int k    = kk32 * 32 + half * 8 + (e & 7) + ((e >> 3) << 4);
    int col  = t * 16 + lr;
    float v  = (col < NCOLS) ? B[(size_t)k * NCOLS + col] : 0.0f;
    P[i] = (__bf16)v;
}

// ---------------------------------------------------------------------------
// Strip-mined WMMA GEMM: C = A @ B (+bias).  A: nrows x 128 f32 row-major
// (nrows % 16 == 0); B pre-packed to bf16 fragments (see pack_B) and staged
// in LDS once per workgroup.  One wave computes a 16-row strip across ALL
// column tiles: A fragment loaded once per 32-wide k-step (4x b128 + cvt),
// then NTC back-to-back v_wmma_f32_16x16x32_bf16 fed by ds_load_b128 pairs.
//   A 16x32 bf16: lane L holds row L%16; K = (L>=16?8:0) + (e&7) + (e>>3)*16
//   C/D f32:      lane L holds col L%16; row = (L>=16?8:0) + vgpr r
// ---------------------------------------------------------------------------
template <int NCOLS>
__global__ void gemm_wmma_bf16(const float* __restrict__ A,
                               const __bf16* __restrict__ Bp,
                               const float* __restrict__ bias,
                               float* __restrict__ C,
                               int nrowtiles) {
    constexpr int NTC  = (NCOLS + 15) / 16;
    constexpr int BELE = 4 * NTC * 32 * 16;           // bf16 elements
    __shared__ __attribute__((aligned(32))) __bf16 sB[BELE];

    // Cooperative LDS fill (16B chunks), no early exit before the barrier.
    {
        const uint4* gsrc = (const uint4*)Bp;
        uint4*       ldst = (uint4*)sB;
        constexpr int NV4 = BELE * 2 / 16;
        for (int i = threadIdx.x; i < NV4; i += blockDim.x) ldst[i] = gsrc[i];
    }
    __syncthreads();

    int tmRaw = blockIdx.x * (blockDim.x >> 5) + (threadIdx.x >> 5);
    bool valid = (tmRaw < nrowtiles);
    int tm = valid ? tmRaw : (nrowtiles - 1);          // clamp; skip stores
    int lane = threadIdx.x & 31;
    int half = lane >> 4;
    int lr   = lane & 15;
    const float* arp = A + (size_t)(tm * 16 + lr) * DMODEL;

    v8f acc[NTC];
#pragma unroll
    for (int t = 0; t < NTC; ++t) acc[t] = (v8f){};

#pragma unroll
    for (int kk32 = 0; kk32 < 4; ++kk32) {
        v16bf a;
        int ka = kk32 * 32 + half * 8;
#pragma unroll
        for (int e = 0; e < 16; ++e)
            a[e] = (__bf16)arp[ka + (e & 7) + ((e >> 3) << 4)];

#pragma unroll
        for (int t = 0; t < NTC; ++t) {
            v16bf b = *(const v16bf*)&sB[((kk32 * NTC + t) * 32 + lane) * 16];
            acc[t] = __builtin_amdgcn_wmma_f32_16x16x32_bf16(
                /*neg_a=*/false, a, /*neg_b=*/false, b,
                /*c_mod=*/(short)0, acc[t], /*reuse_a=*/false, /*reuse_b=*/false);
        }
    }

    if (valid) {
#pragma unroll
        for (int t = 0; t < NTC; ++t) {
            int bc = t * 16 + lr;
            if (bc < NCOLS) {
                float bv = bias ? bias[bc] : 0.0f;
#pragma unroll
                for (int r = 0; r < 8; ++r)
                    C[(size_t)(tm * 16 + half * 8 + r) * NCOLS + bc] = acc[t][r] + bv;
            }
        }
    }
}

// el/er: per (node, head) dot of h[n, head, :] with attn_l / attn_r
__global__ void attn_proj(const float* __restrict__ h,
                          const float* __restrict__ al,
                          const float* __restrict__ ar,
                          float* __restrict__ el, float* __restrict__ er, int N) {
    int i = blockIdx.x * blockDim.x + threadIdx.x;
    if (i >= N * NHEADS) return;
    int n = i >> 2, hd = i & 3;
    const float* hp  = h  + (size_t)n * DMODEL + hd * HIDF;
    const float* alp = al + hd * HIDF;
    const float* arp = ar + hd * HIDF;
    float sl = 0.f, sr = 0.f;
#pragma unroll
    for (int f = 0; f < HIDF; ++f) {
        float v = hp[f];
        sl += v * alp[f];
        sr += v * arp[f];
    }
    el[i] = sl; er[i] = sr;
}

// out <- bias broadcast; softmax state <- identity
__global__ void init_layer(float* __restrict__ out, const float* __restrict__ bias,
                           unsigned* __restrict__ m_u, float* __restrict__ ssum, int N) {
    int i = blockIdx.x * blockDim.x + threadIdx.x;
    if (i < N * DMODEL) out[i] = bias[i & (DMODEL - 1)];
    if (i < N * NHEADS) { m_u[i] = 0u; ssum[i] = 0.f; }
}

__device__ __forceinline__ float lrelu(float x) {
    return x > 0.f ? x : NEG_SLOPE * x;
}

__global__ void edge_max(const int* __restrict__ src, const int* __restrict__ dst,
                         const float* __restrict__ el, const float* __restrict__ er,
                         unsigned* __restrict__ m_u, int E) {
    int e = blockIdx.x * blockDim.x + threadIdx.x;
    if (e >= E) return;
    int s = src[e], d = dst[e];
#pragma unroll
    for (int hd = 0; hd < NHEADS; ++hd) {
        float sc = lrelu(el[s * NHEADS + hd] + er[d * NHEADS + hd]);
        atomicMax(&m_u[d * NHEADS + hd], fenc(sc));
    }
}

__global__ void edge_expsum(const int* __restrict__ src, const int* __restrict__ dst,
                            const float* __restrict__ el, const float* __restrict__ er,
                            const unsigned* __restrict__ m_u, float* __restrict__ ssum, int E) {
    int e = blockIdx.x * blockDim.x + threadIdx.x;
    if (e >= E) return;
    int s = src[e], d = dst[e];
#pragma unroll
    for (int hd = 0; hd < NHEADS; ++hd) {
        float sc = lrelu(el[s * NHEADS + hd] + er[d * NHEADS + hd]);
        float ev = __expf(sc - fdec(m_u[d * NHEADS + hd]));
        atomicAdd(&ssum[d * NHEADS + hd], ev);
    }
}

// One wave per edge; 32 lanes cover 128 channels in 4 head-steps.
// Gather h[src] (L2-resident: 51 MB << 192 MB L2), scatter-add to out[dst].
// Prefetch the next edge's source row (global_prefetch_b8) to hide latency.
__global__ void edge_aggregate(const float* __restrict__ h,
                               const int* __restrict__ src, const int* __restrict__ dst,
                               const float* __restrict__ el, const float* __restrict__ er,
                               const unsigned* __restrict__ m_u, const float* __restrict__ ssum,
                               float* __restrict__ out, int E) {
    int gid  = blockIdx.x * blockDim.x + threadIdx.x;
    int e    = gid >> 5;
    int lane = gid & 31;
    if (e >= E) return;
    int s = src[e], d = dst[e];
    if (e + 1 < E) {
        int sn = src[e + 1];
        __builtin_prefetch(h + (size_t)sn * DMODEL + lane * 4, 0, 1);
    }
    const float* hs = h   + (size_t)s * DMODEL;
    float*       od = out + (size_t)d * DMODEL;
#pragma unroll
    for (int hd = 0; hd < NHEADS; ++hd) {
        float sc = lrelu(el[s * NHEADS + hd] + er[d * NHEADS + hd]);
        float a  = __expf(sc - fdec(m_u[d * NHEADS + hd])) / ssum[d * NHEADS + hd];
        int c = hd * HIDF + lane;
        atomicAdd(&od[c], hs[c] * a);
    }
}

__global__ void elu_inplace(float* __restrict__ x, int n) {
    int i = blockIdx.x * blockDim.x + threadIdx.x;
    if (i < n) {
        float v = x[i];
        x[i] = v > 0.f ? v : (__expf(v) - 1.f);
    }
}

// ---------------------------------------------------------------------------
static void run_gat_layer(const float* xin, const int* src, const int* dst,
                          const float* W, const float* al, const float* ar,
                          const float* bias,
                          float* bufH, float* bufAgg, float* el, float* er,
                          unsigned* m_u, float* ssum, __bf16* packB,
                          int N, int E, hipStream_t stream) {
    int nrowtiles = N / 16;
    int wavesPerBlock = 8;
    int gblocks = (nrowtiles + wavesPerBlock - 1) / wavesPerBlock;

    pack_B<DMODEL><<<(4 * 8 * 512 + 255) / 256, 256, 0, stream>>>(W, packB);
    gemm_wmma_bf16<DMODEL><<<gblocks, wavesPerBlock * 32, 0, stream>>>(
        xin, packB, nullptr, bufH, nrowtiles);

    attn_proj<<<(N * NHEADS + 255) / 256, 256, 0, stream>>>(bufH, al, ar, el, er, N);
    init_layer<<<(N * DMODEL + 255) / 256, 256, 0, stream>>>(bufAgg, bias, m_u, ssum, N);
    edge_max<<<(E + 255) / 256, 256, 0, stream>>>(src, dst, el, er, m_u, E);
    edge_expsum<<<(E + 255) / 256, 256, 0, stream>>>(src, dst, el, er, m_u, ssum, E);
    {
        long long threads = (long long)E * 32;
        int blocks = (int)((threads + 255) / 256);
        edge_aggregate<<<blocks, 256, 0, stream>>>(bufH, src, dst, el, er, m_u, ssum, bufAgg, E);
    }
    elu_inplace<<<(N * DMODEL + 255) / 256, 256, 0, stream>>>(bufAgg, N * DMODEL);
}

extern "C" void kernel_launch(void* const* d_in, const int* in_sizes, int n_in,
                              void* d_out, int out_size, void* d_ws, size_t ws_size,
                              hipStream_t stream) {
    const float* x    = (const float*)d_in[0];
    const int*   src  = (const int*)  d_in[1];
    const int*   dst  = (const int*)  d_in[2];
    const float* W1   = (const float*)d_in[3];
    const float* al1  = (const float*)d_in[4];
    const float* ar1  = (const float*)d_in[5];
    const float* b1   = (const float*)d_in[6];
    const float* W2   = (const float*)d_in[7];
    const float* al2  = (const float*)d_in[8];
    const float* ar2  = (const float*)d_in[9];
    const float* b2   = (const float*)d_in[10];
    const float* Wout = (const float*)d_in[11];
    const float* bout = (const float*)d_in[12];

    int N = in_sizes[0] / DMODEL;   // 100000 (multiple of 16)
    int E = in_sizes[1];            // 1600000

    // Workspace (~109 MB): two N x 128 f32 buffers + softmax state + packed B
    float*    bufH   = (float*)d_ws;
    float*    bufAgg = bufH   + (size_t)N * DMODEL;
    float*    el     = bufAgg + (size_t)N * DMODEL;
    float*    er     = el     + (size_t)N * NHEADS;
    unsigned* m_u    = (unsigned*)(er + (size_t)N * NHEADS);
    float*    ssum   = (float*)(m_u + (size_t)N * NHEADS);
    __bf16*   packB  = (__bf16*)(ssum + (size_t)N * NHEADS);   // <= 32 KB

    // Layer 1: x -> bufAgg (elu'd)
    run_gat_layer(x, src, dst, W1, al1, ar1, b1,
                  bufH, bufAgg, el, er, m_u, ssum, packB, N, E, stream);
    // Layer 2: bufAgg -> bufH (gemm), aggregate back into bufAgg (elu'd)
    run_gat_layer(bufAgg, src, dst, W2, al2, ar2, b2,
                  bufH, bufAgg, el, er, m_u, ssum, packB, N, E, stream);

    // Output head: d_out = elu(h2) @ Wout + bout   (N x 128 @ 128 x 40)
    {
        int nrowtiles = N / 16;
        int wavesPerBlock = 8;
        int gblocks = (nrowtiles + wavesPerBlock - 1) / wavesPerBlock;
        pack_B<40><<<(4 * 3 * 512 + 255) / 256, 256, 0, stream>>>(Wout, packB);
        gemm_wmma_bf16<40><<<gblocks, wavesPerBlock * 32, 0, stream>>>(
            bufAgg, packB, bout, (float*)d_out, nrowtiles);
    }
    (void)n_in; (void)out_size; (void)ws_size;
}
